// GraphConvolutionLayer_63041529970791
// MI455X (gfx1250) — compile-verified, matrix operand order checked by reading, not run on hardware
//
#include <hip/hip_runtime.h>
#include <hip/hip_bf16.h>

typedef __attribute__((ext_vector_type(16))) _Float16 v16h;
typedef __attribute__((ext_vector_type(8)))  float    v8f;

#define B_     4
#define T_     12
#define N_     2000
#define K_     10
#define F_     64
#define H_     8
#define BT_    (B_ * T_)            // 48
#define ROWS_  (BT_ * H_)           // 384 rows of A per n
#define LDSTR  65                   // padded row stride (bank-conflict free)
#define ALPHA_ 0.2f
#define SIG2I  (1.0f / 36.0f)

__global__ __launch_bounds__(256)
void graphconv_wmma_kernel(const float* __restrict__ x,
                           const float* __restrict__ dist,
                           const float* __restrict__ W,
                           const float* __restrict__ bias,
                           const int*   __restrict__ nidx,
                           float*       __restrict__ out)
{
    __shared__ float aggS[ROWS_ * LDSTR];   // 384*65*4 = 97.5 KB
    __shared__ float Wsh [F_ * LDSTR];      // 64*65*4  = 16.3 KB
    __shared__ float wgt [K_ * H_];         // 80 kernel weights
    __shared__ int   idxS[K_ + 2];

    const int n   = blockIdx.x;
    const int tid = threadIdx.x;

    // ---------------- Stage 0: weights, indices, W -> LDS ----------------
    if (tid < K_ * H_) {
        const int k = tid >> 3, h = tid & 7;
        const float d   = dist[n * K_ + k];
        const float lam = (float)(h + 1) * 0.125f;
        wgt[tid] = __expf(-(d * d) * lam * SIG2I);
    }
    if (tid < K_) idxS[tid] = nidx[n * K_ + tid];
    #pragma unroll
    for (int i = tid; i < F_ * F_; i += 256)
        Wsh[(i >> 6) * LDSTR + (i & 63)] = W[i];
    __syncthreads();

    // ---------------- Stage 1: gather + per-head aggregation -------------
    // thread -> (f = tid&63), bt-group = tid>>6 ; x loaded once, reused 8x.
    // x (24.6 MB) stays L2-resident: loads use default RT temporal policy.
    {
        const int f = tid & 63;
        const int g = tid >> 6;
        int nk[K_];
        #pragma unroll
        for (int k = 0; k < K_; ++k) nk[k] = idxS[k];

        for (int bt = g; bt < BT_; bt += 4) {
            float xv[K_];
            #pragma unroll
            for (int k = 0; k < K_; ++k)
                xv[k] = x[((size_t)bt * N_ + nk[k]) * F_ + f];
            #pragma unroll
            for (int h = 0; h < H_; ++h) {
                float acc = 0.0f;
                #pragma unroll
                for (int k = 0; k < K_; ++k)
                    acc = fmaf(wgt[k * H_ + h], xv[k], acc);
                aggS[(bt * H_ + h) * LDSTR + f] = acc;
            }
        }
    }
    __syncthreads();

    // ---------------- Stage 2: blend + WMMA GEMM (A[384x64] * W^T) -------
    const int wave     = tid >> 5;         // 8 waves
    const int lane     = tid & 31;
    const int laneHalf = lane & 15;
    const int hiHalf   = (lane >> 4) & 1;  // lanes 16..31

    // B fragments: B[k][f'] = W[f', k]; row f' of W is contiguous in LDS.
    // 16-bit fragment K-map: VGPR j holds K = 2*(j&3) + 16*(j>=4) + 8*hiHalf.
    v16h bfrag[4][2];
    float biasv[4];
    #pragma unroll
    for (int ct = 0; ct < 4; ++ct) {
        const int fp = ct * 16 + laneHalf;
        biasv[ct] = bias[fp];
        const float* wrow = &Wsh[fp * LDSTR];
        #pragma unroll
        for (int kc = 0; kc < 2; ++kc) {
            #pragma unroll
            for (int j = 0; j < 8; ++j) {
                const int k = kc * 32 + ((j & 3) << 1) + ((j >> 2) << 4) + (hiHalf << 3);
                bfrag[ct][kc][2 * j]     = (_Float16)wrow[k];
                bfrag[ct][kc][2 * j + 1] = (_Float16)wrow[k + 1];
            }
        }
    }

    // 24 row-tiles / 8 waves = 3 tiles each, no divergence around WMMA.
    #pragma unroll
    for (int i = 0; i < 3; ++i) {
        const int rt  = wave * 3 + i;
        const int row = rt * 16 + laneHalf;
        const int bt  = row >> 3;
        const int t   = bt % T_;
        const float* cur  = &aggS[row * LDSTR];
        const float* prev = &aggS[(row - H_) * LDSTR]; // (bt-1, same h)
        const float blendf = (t == 0) ? 0.0f : ALPHA_; // fused temporal blend

        v16h afrag[2];
        #pragma unroll
        for (int kc = 0; kc < 2; ++kc) {
            #pragma unroll
            for (int j = 0; j < 8; ++j) {
                const int k = kc * 32 + ((j & 3) << 1) + ((j >> 2) << 4) + (hiHalf << 3);
                const float v0 = cur[k]     + blendf * (prev[k]     - cur[k]);
                const float v1 = cur[k + 1] + blendf * (prev[k + 1] - cur[k + 1]);
                afrag[kc][2 * j]     = (_Float16)v0;
                afrag[kc][2 * j + 1] = (_Float16)v1;
            }
        }

        #pragma unroll
        for (int ct = 0; ct < 4; ++ct) {
            v8f c = {};
            c = __builtin_amdgcn_wmma_f32_16x16x32_f16(
                    false, afrag[0], false, bfrag[ct][0], (short)0, c, false, false);
            c = __builtin_amdgcn_wmma_f32_16x16x32_f16(
                    false, afrag[1], false, bfrag[ct][1], (short)0, c, false, false);

            // D layout: VGPR r -> M = r + 8*hiHalf, lane%16 -> N. Fuse bias+ReLU.
            // Output stream is 196.6 MB (~= L2 size): store NON-TEMPORAL so the
            // write stream does not evict the L2-resident, 10x-reused x tensor.
            const float bb = biasv[ct];
            #pragma unroll
            for (int r = 0; r < 8; ++r) {
                const int grow = rt * 16 + r + (hiHalf << 3);
                const int obt  = grow >> 3;
                const int oh   = grow & 7;
                const size_t off = (((size_t)obt * N_ + n) * H_ + oh) * F_
                                   + ct * 16 + laneHalf;
                __builtin_nontemporal_store(fmaxf(c[r] + bb, 0.0f), &out[off]);
            }
        }
    }
}

extern "C" void kernel_launch(void* const* d_in, const int* in_sizes, int n_in,
                              void* d_out, int out_size, void* d_ws, size_t ws_size,
                              hipStream_t stream)
{
    const float* x    = (const float*)d_in[0];
    const float* dist = (const float*)d_in[1];
    const float* W    = (const float*)d_in[2];
    const float* bias = (const float*)d_in[3];
    const int*   nidx = (const int*)d_in[4];
    float*       out  = (float*)d_out;

    graphconv_wmma_kernel<<<dim3(N_), dim3(256), 0, stream>>>(
        x, dist, W, bias, nidx, out);
}
